// NeuralNetwork_12034498363773
// MI455X (gfx1250) — compile-verified, matrix operand order checked by reading, not run on hardware
//
#include <hip/hip_runtime.h>

typedef __attribute__((ext_vector_type(16))) _Float16 v16h;
typedef __attribute__((ext_vector_type(8)))  _Float16 v8h;
typedef __attribute__((ext_vector_type(4)))  _Float16 v4h;
typedef __attribute__((ext_vector_type(8)))  float    v8f;
typedef __attribute__((ext_vector_type(4)))  float    v4f;
typedef __attribute__((ext_vector_type(4)))  int      v4i;

#define N_INPUT   256
#define N_HIDDEN  2048
#define BATCH     131072
#define M_BLOCK   128                 // batch rows per workgroup (8 waves x 16 rows)
#define TILE_N    16
#define N_TILES   (N_HIDDEN / TILE_N) // 128
#define PITCH     264                 // padded LDS row pitch (f16) -> conflict-light 16B reads

__device__ __forceinline__ float fast_tanh(float x) {
#if __has_builtin(__builtin_amdgcn_tanhf)
    return __builtin_amdgcn_tanhf(x);      // gfx1250 v_tanh_f32 (confirmed in round-1 asm)
#elif __has_builtin(__builtin_amdgcn_tanh_f32)
    return __builtin_amdgcn_tanh_f32(x);
#else
    return tanhf(x);
#endif
}

// ---- async global->LDS staging (CDNA5 GLOBAL_LOAD_ASYNC_TO_LDS_B128, ASYNCcnt) ----
// Param 1 is 'AS(1) v4i*' per the round-2 diagnostic; AS(3) v4i* for the LDS side.
#if __has_builtin(__builtin_amdgcn_global_load_async_to_lds_b128)
#define HAVE_ASYNC_LDS 1
__device__ __forceinline__ void async_copy_b128(const _Float16* g, _Float16* l) {
    __builtin_amdgcn_global_load_async_to_lds_b128(
        (__attribute__((address_space(1))) v4i*)g,
        (__attribute__((address_space(3))) v4i*)l,
        /*offset=*/0, /*cpol=*/0);
}
#else
#define HAVE_ASYNC_LDS 0
#endif

__device__ __forceinline__ void wait_async0() {
#if HAVE_ASYNC_LDS
#if __has_builtin(__builtin_amdgcn_s_wait_asynccnt)
    __builtin_amdgcn_s_wait_asynccnt(0);
#else
    asm volatile("s_wait_asynccnt 0x0" ::: "memory");
#endif
#endif
}

// ---- pre-pass: convert W1 (2048x256 fp32, row-major) to fp16 in workspace ----
__global__ void cvt_w1_kernel(const float* __restrict__ w1, _Float16* __restrict__ w1h) {
    int i = (blockIdx.x * 256 + threadIdx.x) * 4;   // 512 blocks x 256 thr x 4 = 524288
    v4f f = *(const v4f*)(w1 + i);
    v4h h;
    h[0] = (_Float16)f[0]; h[1] = (_Float16)f[1];
    h[2] = (_Float16)f[2]; h[3] = (_Float16)f[3];
    *(v4h*)(w1h + i) = h;
}

// ---- fused MLP: lyap = tanh(tanh(x@W1^T + b1)@W2^T + b2); ctrl = clip(x@Wc, -1, 1) ----
__global__ __launch_bounds__(256)
void fused_mlp_kernel(const float* __restrict__ x,
                      const _Float16* __restrict__ w1h,
                      const float* __restrict__ b1,
                      const float* __restrict__ w2,
                      const float* __restrict__ b2,
                      const float* __restrict__ wc,
                      float* __restrict__ out) {
    __shared__ _Float16 lds[2][TILE_N * PITCH];

    const int tid  = threadIdx.x;
    const int wave = tid >> 5;
    const int lane = tid & 31;
    const int nl   = lane & 15;       // N index within tile (C layout), row within A
    const int hi   = lane >> 4;       // half-wave select
    const int rowBase = blockIdx.x * M_BLOCK + wave * 16;

    // ---------------- control head: out[BATCH + row] = clip(x_row . Wc) ----------------
    float wcl[8];
    {
        const float* wp = wc + lane * 8;
        #pragma unroll
        for (int i = 0; i < 8; ++i) wcl[i] = wp[i];
    }
    for (int m = 0; m < 16; ++m) {
        const float* xr = x + (size_t)(rowBase + m) * N_INPUT + lane * 8;
        float p = 0.f;
        #pragma unroll
        for (int i = 0; i < 8; ++i) p = fmaf(xr[i], wcl[i], p);
        #pragma unroll
        for (int mask = 16; mask >= 1; mask >>= 1) p += __shfl_xor(p, mask, 32);
        if (lane == 0) {
            out[BATCH + rowBase + m] = fminf(1.0f, fmaxf(-1.0f, p));
        }
    }

    // ---------------- load + convert this wave's A fragments (16 rows x 256 K) ----------------
    // 16-bit A 16x32 layout: lanes 0-15 (row m=lane) hold K = [32s+0,+8) and [32s+16,+8);
    // lanes 16-31 (row m=lane-16) hold K = [32s+8,+8) and [32s+24,+8).
    const int csub = hi ? 8 : 0;
    const float* xrow = x + (size_t)(rowBase + nl) * N_INPUT;
    v16h a[8];
    #pragma unroll
    for (int s = 0; s < 8; ++s) {
        const float* p0 = xrow + 32 * s + csub;
        v4f f0 = *(const v4f*)(p0);
        v4f f1 = *(const v4f*)(p0 + 4);
        v4f f2 = *(const v4f*)(p0 + 16);
        v4f f3 = *(const v4f*)(p0 + 20);
        v16h av;
        #pragma unroll
        for (int i = 0; i < 4; ++i) {
            av[i]      = (_Float16)f0[i];
            av[4 + i]  = (_Float16)f1[i];
            av[8 + i]  = (_Float16)f2[i];
            av[12 + i] = (_Float16)f3[i];
        }
        a[s] = av;
    }

    // ---------------- tile staging: 16x256 f16 (8 KB), 32 B per thread ----------------
    const int sr   = tid >> 4;          // dest row in tile
    const int scol = (tid & 15) * 16;   // dest col (f16), 32B-aligned chunks
    auto stage = [&](int buf, int tile) {
        const _Float16* src = w1h + (size_t)tile * (TILE_N * N_INPUT) + tid * 16;
        _Float16* dst = &lds[buf][sr * PITCH + scol];
#if HAVE_ASYNC_LDS
        async_copy_b128(src,     dst);
        async_copy_b128(src + 8, dst + 8);
#else
        *(v4i*)(dst)     = *(const v4i*)(src);
        *(v4i*)(dst + 8) = *(const v4i*)(src + 8);
#endif
    };

    // B fragment load from LDS (32x16 KxN tile, k-step s): column n = row n of W1 tile.
    auto loadB = [&](const _Float16* br, int s) -> v16h {
        v8h blo = *(const v8h*)(br + 32 * s + csub);
        v8h bhi = *(const v8h*)(br + 32 * s + csub + 16);
        return __builtin_shufflevector(blo, bhi,
                 0, 1, 2, 3, 4, 5, 6, 7, 8, 9, 10, 11, 12, 13, 14, 15);
    };

    float acc[8] = {0.f, 0.f, 0.f, 0.f, 0.f, 0.f, 0.f, 0.f};

    stage(0, 0);
    wait_async0();
    __syncthreads();

    for (int t = 0; t < N_TILES; ++t) {
        const int buf = t & 1;
        if (t + 1 < N_TILES) stage(buf ^ 1, t + 1);   // overlap copy with WMMA chain

        const _Float16* br = &lds[buf][nl * PITCH];

        // two independent accumulator chains -> overlapped WMMA issue
        v8f c0 = {}, c1 = {};
        #pragma unroll
        for (int s = 0; s < 8; s += 2) {
            v16h bv0 = loadB(br, s);
            v16h bv1 = loadB(br, s + 1);
            c0 = __builtin_amdgcn_wmma_f32_16x16x32_f16(
                     false, a[s],     false, bv0, (short)0, c0, false, false);
            c1 = __builtin_amdgcn_wmma_f32_16x16x32_f16(
                     false, a[s + 1], false, bv1, (short)0, c1, false, false);
        }
        v8f cc = c0 + c1;

        const int ng = t * 16 + nl;
        const float b1n = b1[ng];
        const float w2n = w2[ng];
        // C layout: lane<16 -> (M=v, N=nl); lane>=16 -> (M=v+8, N=nl)
        #pragma unroll
        for (int v = 0; v < 8; ++v)
            acc[v] = fmaf(fast_tanh(cc[v] + b1n), w2n, acc[v]);

        wait_async0();       // my async writes for buf^1 have landed
        __syncthreads();     // everyone's landed; buf free for next stage
    }

    // ---------------- reduce over the 16 N-lanes of each half and finish ----------------
    const float b2v = b2[0];
    #pragma unroll
    for (int v = 0; v < 8; ++v) {
        float r = acc[v];
        r += __shfl_xor(r, 1, 32);
        r += __shfl_xor(r, 2, 32);
        r += __shfl_xor(r, 4, 32);
        r += __shfl_xor(r, 8, 32);
        if (nl == 0) {                                   // lanes 0 and 16
            out[rowBase + hi * 8 + v] = fast_tanh(r + b2v);
        }
    }
}

extern "C" void kernel_launch(void* const* d_in, const int* in_sizes, int n_in,
                              void* d_out, int out_size, void* d_ws, size_t ws_size,
                              hipStream_t stream) {
    const float* x  = (const float*)d_in[0];  // [131072, 256]
    const float* w1 = (const float*)d_in[1];  // [2048, 256]
    const float* b1 = (const float*)d_in[2];  // [2048]
    const float* w2 = (const float*)d_in[3];  // [1, 2048]
    const float* b2 = (const float*)d_in[4];  // [1]
    const float* wc = (const float*)d_in[5];  // [1, 256]
    float* out = (float*)d_out;               // [131072 lyap | 131072 ctrl]

    _Float16* w1h = (_Float16*)d_ws;          // 1 MB fp16 copy of W1

    cvt_w1_kernel<<<(N_HIDDEN * N_INPUT) / (256 * 4), 256, 0, stream>>>(w1, w1h);
    fused_mlp_kernel<<<BATCH / M_BLOCK, 256, 0, stream>>>(x, w1h, b1, w2, b2, wc, out);
}
